// TritonTrittention_47588237639881
// MI455X (gfx1250) — matrix operand order repaired and checked by reading, not compiled, not run on hardware
//
#include <hip/hip_runtime.h>
#include <hip/hip_bf16.h>

typedef _Float16 h4   __attribute__((ext_vector_type(4)));
typedef _Float16 h8   __attribute__((ext_vector_type(8)));
typedef _Float16 v16h __attribute__((ext_vector_type(16)));
typedef float    v8f  __attribute__((ext_vector_type(8)));

#define T_SCALE 0.125f
#define LOG2E   1.4426950408889634f
#define SEQ 256
#define HDIM 64

// One-instruction lane-XOR shuffle: ds_swizzle_b32, group-of-32 xor pattern.
template <int MASK>
__device__ __forceinline__ float swzx(float x) {
    return __int_as_float(
        __builtin_amdgcn_ds_swizzle(__float_as_int(x), (MASK << 10) | 31));
}

// Dynamic LDS: W (q ⊙ K1) f16 row-major [256][64], then K2 f16 row-major [256][64]
extern __shared__ _Float16 dynLds[];

__global__ __launch_bounds__(256) void tritt_kernel(
    const float* __restrict__ Qg,  const float* __restrict__ K1g,
    const float* __restrict__ K2g, const float* __restrict__ V1g,
    const float* __restrict__ V2g, float* __restrict__ Og,
    float* __restrict__ Mg)
{
    _Float16* sW  = dynLds;                 // 256*64 f16 = 32 KB
    _Float16* sK2 = dynLds + SEQ * HDIM;    // 256*64 f16 = 32 KB

    __shared__ float sQ[HDIM];
    __shared__ float sR[SEQ];        // row sums of exp(s - m)
    __shared__ float sC[SEQ];        // col sums of exp(s - m)
    __shared__ float sMax[8];
    __shared__ float sZ;
    __shared__ float sO[16][HDIM];   // epilogue partials (4 KB)

    const int tid  = threadIdx.x;
    const int lane = tid & 31;
    const int wave = tid >> 5;
    const int l15  = lane & 15;
    const int hi   = lane >> 4;           // which 16-lane half
    const int qidx = blockIdx.x & (SEQ - 1);
    const int h    = blockIdx.x >> 8;
    const size_t headOff = (size_t)h * SEQ * HDIM;

    // ---- init small LDS ----
    if (tid < HDIM) sQ[tid] = Qg[headOff + (size_t)qidx * HDIM + tid];
    sC[tid] = 0.0f;
    if (tid == 0) sZ = 0.0f;
    __syncthreads();

    // ---- stage W = Q(q) ⊙ K1 and K2 into LDS as f16 (b128 loads, b64 stores) ----
    {
        const float4* K1v = (const float4*)(K1g + headOff);
        const float4* K2v = (const float4*)(K2g + headOff);
        #pragma unroll 1
        for (int i4 = tid; i4 < SEQ * HDIM / 4; i4 += 256) {
            const int d = (i4 << 2) & (HDIM - 1);
            const float4 a = K1v[i4];
            const float4 b = K2v[i4];
            h4 w, k;
            w[0] = (_Float16)(sQ[d + 0] * a.x);
            w[1] = (_Float16)(sQ[d + 1] * a.y);
            w[2] = (_Float16)(sQ[d + 2] * a.z);
            w[3] = (_Float16)(sQ[d + 3] * a.w);
            k[0] = (_Float16)b.x;  k[1] = (_Float16)b.y;
            k[2] = (_Float16)b.z;  k[3] = (_Float16)b.w;
            *(h4*)&sW[i4 << 2]  = w;
            *(h4*)&sK2[i4 << 2] = k;
        }
    }
    __syncthreads();

    // A fragment (16x32 f16): lane half selects K {0..7,16..23} vs {8..15,24..31}
    auto loadA = [&](int ib, int kc) -> v16h {
        const int row  = ib * 16 + l15;
        const int base = row * HDIM + kc * 32 + hi * 8;
        h8 lo = *(const h8*)&sW[base];
        h8 hv = *(const h8*)&sW[base + 16];
        return __builtin_shufflevector(lo, hv, 0,1,2,3,4,5,6,7,8,9,10,11,12,13,14,15);
    };
    // B fragment (32x16 f16): lane = column j, lane half selects K {0..7,16..23} vs {8..15,24..31}
    auto loadB = [&](int jb, int kc) -> v16h {
        const int col  = jb * 16 + l15;
        const int base = col * HDIM + kc * 32 + hi * 8;
        h8 lo = *(const h8*)&sK2[base];
        h8 hv = *(const h8*)&sK2[base + 16];
        return __builtin_shufflevector(lo, hv, 0,1,2,3,4,5,6,7,8,9,10,11,12,13,14,15);
    };

    // Both i-blocks' A fragments live in registers for the whole kernel.
    const int ib0 = wave, ib1 = wave + 8;
    const v16h a00 = loadA(ib0, 0), a01 = loadA(ib0, 1);
    const v16h a10 = loadA(ib1, 0), a11 = loadA(ib1, 1);

    // ================= pass 1: global max of raw scores =================
    float lmax = -3.0e38f;
    #pragma unroll 1
    for (int jb = 0; jb < 16; ++jb) {
        const v16h b0 = loadB(jb, 0), b1 = loadB(jb, 1);
        v8f c0 = {}, c1 = {};
        c0 = __builtin_amdgcn_wmma_f32_16x16x32_f16(false, a00, false, b0, (short)0, c0, false, false);
        c1 = __builtin_amdgcn_wmma_f32_16x16x32_f16(false, a10, false, b0, (short)0, c1, false, false);
        c0 = __builtin_amdgcn_wmma_f32_16x16x32_f16(false, a01, false, b1, (short)0, c0, false, false);
        c1 = __builtin_amdgcn_wmma_f32_16x16x32_f16(false, a11, false, b1, (short)0, c1, false, false);
        #pragma unroll
        for (int v = 0; v < 8; ++v) lmax = fmaxf(lmax, fmaxf(c0[v], c1[v]));
    }
    lmax *= T_SCALE;                        // SCALE > 0: scale once after the max
    lmax = fmaxf(lmax, swzx<16>(lmax));
    lmax = fmaxf(lmax, swzx<8>(lmax));
    lmax = fmaxf(lmax, swzx<4>(lmax));
    lmax = fmaxf(lmax, swzx<2>(lmax));
    lmax = fmaxf(lmax, swzx<1>(lmax));
    if (lane == 0) sMax[wave] = lmax;
    __syncthreads();
    float gmax = sMax[0];
    #pragma unroll
    for (int w = 1; w < 8; ++w) gmax = fmaxf(gmax, sMax[w]);

    // exp(c*S - g) == exp2(c*kk + ng2): one fma + native v_exp_f32
    const float kk  = T_SCALE * LOG2E;
    const float ng2 = -gmax * LOG2E;

    // ======== pass 2: exp, row sums (regs) + col sums (LDS atomics) ========
    float racc0[8], racc1[8];
    #pragma unroll
    for (int v = 0; v < 8; ++v) { racc0[v] = 0.0f; racc1[v] = 0.0f; }

    #pragma unroll 1
    for (int jb = 0; jb < 16; ++jb) {
        const v16h b0 = loadB(jb, 0), b1 = loadB(jb, 1);
        v8f c0 = {}, c1 = {};
        c0 = __builtin_amdgcn_wmma_f32_16x16x32_f16(false, a00, false, b0, (short)0, c0, false, false);
        c1 = __builtin_amdgcn_wmma_f32_16x16x32_f16(false, a10, false, b0, (short)0, c1, false, false);
        c0 = __builtin_amdgcn_wmma_f32_16x16x32_f16(false, a01, false, b1, (short)0, c0, false, false);
        c1 = __builtin_amdgcn_wmma_f32_16x16x32_f16(false, a11, false, b1, (short)0, c1, false, false);

        float cs = 0.0f;
        #pragma unroll
        for (int v = 0; v < 8; ++v) {
            float p0 = __builtin_amdgcn_exp2f(fmaf(c0[v], kk, ng2));
            float p1 = __builtin_amdgcn_exp2f(fmaf(c1[v], kk, ng2));
            racc0[v] += p0;
            racc1[v] += p1;
            cs += p0 + p1;                  // both i-blocks hit the same column
        }
        cs += swzx<16>(cs);                 // combine both row halves
        if (lane < 16) atomicAdd(&sC[jb * 16 + lane], cs);  // 16 banks, no conflict
    }

    // reduce rows over the 16 lanes of each half (columns of this wave)
    #pragma unroll
    for (int v = 0; v < 8; ++v) {
        racc0[v] += swzx<8>(racc0[v]);
        racc0[v] += swzx<4>(racc0[v]);
        racc0[v] += swzx<2>(racc0[v]);
        racc0[v] += swzx<1>(racc0[v]);
        racc1[v] += swzx<8>(racc1[v]);
        racc1[v] += swzx<4>(racc1[v]);
        racc1[v] += swzx<2>(racc1[v]);
        racc1[v] += swzx<1>(racc1[v]);
    }
    if (l15 == 0) {                  // lane 0 -> rows +0..7, lane 16 -> +8..15
        #pragma unroll
        for (int v = 0; v < 8; ++v) {
            sR[ib0 * 16 + hi * 8 + v] = racc0[v];
            sR[ib1 * 16 + hi * 8 + v] = racc1[v];
        }
    }
    __syncthreads();

    // ================= Z = sum of row sums =================
    {
        float z = sR[tid];
        z += swzx<16>(z);
        z += swzx<8>(z);
        z += swzx<4>(z);
        z += swzx<2>(z);
        z += swzx<1>(z);
        if (lane == 0) atomicAdd(&sZ, z);
    }
    __syncthreads();
    const float Z = sZ;

    // ============ epilogue: O = (r·V1 + c·V2)/Z, float4 vectorized ============
    {
        const int g  = tid >> 4;          // 16 groups of 16 rows
        const int dd = (tid & 15) << 2;   // 4 consecutive d per thread
        const float4* V1v = (const float4*)(V1g + headOff);
        const float4* V2v = (const float4*)(V2g + headOff);
        float ax = 0.0f, ay = 0.0f, az = 0.0f, aw = 0.0f;
        #pragma unroll 4
        for (int i = 0; i < 16; ++i) {
            const int r  = g * 16 + i;
            const int o4 = (r * HDIM + dd) >> 2;
            const float4 v1 = V1v[o4];
            const float4 v2 = V2v[o4];
            const float rr = sR[r], cc = sC[r];
            ax += rr * v1.x + cc * v2.x;
            ay += rr * v1.y + cc * v2.y;
            az += rr * v1.z + cc * v2.z;
            aw += rr * v1.w + cc * v2.w;
        }
        float4 acc = { ax, ay, az, aw };
        *(float4*)&sO[g][dd] = acc;       // stride-4 banks: conflict-free
    }
    __syncthreads();
    if (tid < HDIM) {
        float o = 0.0f;
        #pragma unroll
        for (int g = 0; g < 16; ++g) o += sO[g][tid];
        Og[((size_t)h * SEQ + qidx) * HDIM + tid] = o / Z;
    }
    if (tid == 0) Mg[h * SEQ + qidx] = gmax + logf(Z);
}

extern "C" void kernel_launch(void* const* d_in, const int* in_sizes, int n_in,
                              void* d_out, int out_size, void* d_ws, size_t ws_size,
                              hipStream_t stream) {
    const float* Qg  = (const float*)d_in[0];
    const float* K1g = (const float*)d_in[1];
    const float* K2g = (const float*)d_in[2];
    const float* V1g = (const float*)d_in[3];
    const float* V2g = (const float*)d_in[4];

    float* Og = (float*)d_out;                 // [1,8,256,64]
    float* Mg = Og + (size_t)8 * 256 * 64;     // [1,8,256]

    const size_t shmem = (size_t)2 * SEQ * HDIM * sizeof(unsigned short); // 64 KB f16
    tritt_kernel<<<dim3(8 * SEQ), dim3(256), shmem, stream>>>(Qg, K1g, K2g, V1g, V2g, Og, Mg);
}
